// MultiHeadAttention_9466107920789
// MI455X (gfx1250) — matrix operand order repaired and checked by reading, not compiled, not run on hardware
//
#include <hip/hip_runtime.h>
#include <hip/hip_bf16.h>
#include <stdint.h>

// MI455X (gfx1250) multi-head-attention forward, bf16 WMMA pipeline with
// async-DMA (ASYNCcnt) LDS staging and double buffering.
//  K0: fp32 -> bf16 converts (q,k,v, Wq,Wk,Wv,Wo)
//  K1: qh = q@Wq^T+bq, kh = k@Wk^T+bk (bf16); vhT = (v@Wv^T+bv)^T (bf16)
//  K2: scores = (qh @ kh^T) / 8   (fp32, 64MB, L2-resident)
//  K3: row softmax -> attn (bf16)
//  K4: out = attn @ vh, stored with the reference's head-permutation (bf16)
//  K5: y = outPerm @ Wo^T + bo  (fp32 -> d_out)

typedef __bf16 bf16_t;
typedef __attribute__((ext_vector_type(16))) __bf16 v16bf;
typedef __attribute__((ext_vector_type(8)))  __bf16 v8bf;
typedef __attribute__((ext_vector_type(8)))  float  v8f;

// LDS tile geometry: rows padded to 80B (64B data + 16B pad) => conflict-free
// ds_load_b128 fragment reads (bank stride 20, all 32 lanes disjoint).
#define ROWB      80
#define A_TILE_B  (128 * ROWB)            // 10240 B
#define B_TILE_B  (256 * ROWB)            // 20480 B
#define BUF_B     (A_TILE_B + B_TILE_B)   // 30720 B per buffer
#define LDS_BYTES (2 * BUF_B)             // 61440 B, double buffered

// ---------------------------------------------------------------- converts
__global__ __launch_bounds__(256) void cvt_f32_to_bf16(const float* __restrict__ s,
                                                       bf16_t* __restrict__ d, int n) {
  int i = blockIdx.x * blockDim.x + threadIdx.x;
  int stride = gridDim.x * blockDim.x;
  for (; i < n; i += stride) d[i] = (bf16_t)s[i];
}

// ------------------------------------------------- async global->LDS staging
// Per-lane 16B chunk: row = c>>2, seg = c&3. GVS addressing: 64-bit SGPR base
// + 32-bit VGPR byte offset. Tracked with ASYNCcnt. NROWS*4 is an exact
// multiple of 256 => compile-time trip count, branch-free, EXEC untouched.
template<int NROWS>
static __device__ __forceinline__ void issue_tile_async(
    const bf16_t* __restrict__ base, int lda, int kbyte,
    uint32_t lds_tile_base, int tid) {
  constexpr int ITERS = (NROWS * 4) / 256;
  #pragma unroll
  for (int i = 0; i < ITERS; ++i) {
    const int c = tid + i * 256;
    const int row = c >> 2, seg = c & 3;
    const uint32_t goff = (uint32_t)(row * (lda * 2) + kbyte + seg * 16);
    const uint32_t loff = lds_tile_base + (uint32_t)(row * ROWB + seg * 16);
    asm volatile("global_load_async_to_lds_b128 %0, %1, %2"
                 :: "v"(loff), "v"(goff), "s"(base) : "memory");
  }
}

static __device__ __forceinline__ void wait_async_zero() {
  asm volatile("s_wait_asynccnt 0x0" ::: "memory");
}

// ------------------------------------------------- LDS fragment reads
// A-matrix 16x32 bf16: lanes 0-15: M=lane, elems[0..15]=K 0..15;
// lanes 16-31: same M, K 16..31 (ascending).
static __device__ __forceinline__ v16bf lds_a_frag(const char* __restrict__ smA,
                                                   int localRow, int koffA) {
  const char* p = smA + (size_t)localRow * ROWB + koffA * 2;
  v8bf lo = *(const v8bf*)(p);        // K koffA+0..7
  v8bf hi = *(const v8bf*)(p + 16);   // K koffA+8..15
  return __builtin_shufflevector(lo, hi, 0,1,2,3,4,5,6,7,8,9,10,11,12,13,14,15);
}

// B-matrix 32x16 bf16: lane = column (&15); lanes 0-15 hold K=0..15, lanes
// 16-31 hold K=16..31 (ascending). NT: column n == row n of row-major operand.
static __device__ __forceinline__ v16bf lds_b_frag(const char* __restrict__ smB,
                                                   int localRow, int koffB) {
  const char* p = smB + (size_t)localRow * ROWB + koffB * 2;
  v8bf lo = *(const v8bf*)(p);        // K koffB+0..7
  v8bf hi = *(const v8bf*)(p + 16);   // K koffB+8..15
  return __builtin_shufflevector(lo, hi, 0,1,2,3,4,5,6,7,8,9,10,11,12,13,14,15);
}

// ---------------------------------------------------------------- NT GEMM
// C[M,N] = scale * (A[M,K] @ B[N,K]^T) (+ bias[N]); batched via blockIdx.z.
// Workgroup tile 128x256 (8 waves of 64x64), K-step 32, async double buffer.
// STORE_MODE 0: row-major. 1: vhT scatter [b][d][s] (S=2048, D*S=2^21).
// 2: reference head-permutation scatter.
template<int STORE_MODE, bool OUT_BF16, bool BIAS>
__global__ __launch_bounds__(256, 2) void wmma_nt_gemm(
    const bf16_t* __restrict__ A, const bf16_t* __restrict__ Bm,
    void* __restrict__ Cout, const float* __restrict__ bias, float scale,
    int K, int lda, int ldb, int ldc,
    long aStride, long bStride, long cStride)
{
  __shared__ __align__(128) char lds[LDS_BYTES];

  const int tid  = threadIdx.x;
  const int lane = tid & 31;
  const int wave = tid >> 5;
  const int z    = blockIdx.z;

  const int blockM = blockIdx.y * 128;
  const int blockN = blockIdx.x * 256;
  const bf16_t* Ablk = A  + (size_t)z * aStride + (size_t)blockM * lda;
  const bf16_t* Bblk = Bm + (size_t)z * bStride + (size_t)blockN * ldb;

  const uint32_t ldsb = (uint32_t)(uintptr_t)(void*)lds;

  // prologue: stage K-step 0 into buffer 0
  issue_tile_async<128>(Ablk, lda, 0, ldsb, tid);
  issue_tile_async<256>(Bblk, ldb, 0, ldsb + A_TILE_B, tid);

  const int waveMl = (wave >> 2) * 64;          // local within block
  const int waveNl = (wave &  3) * 64;
  const int mrow   = waveMl + (lane & 15);
  const int nrow   = waveNl + (lane & 15);
  const int koffA  = (lane >> 4) << 4;          // 0 or 16 (elements)
  const int koffB  = (lane >> 4) << 4;          // 0 or 16 (elements)

  const v8f vzero = {0.f,0.f,0.f,0.f,0.f,0.f,0.f,0.f};
  v8f acc[4][4];
  #pragma unroll
  for (int mi = 0; mi < 4; ++mi)
    #pragma unroll
    for (int ni = 0; ni < 4; ++ni) acc[mi][ni] = vzero;

  const int nsteps = K >> 5;
  for (int it = 0; it < nsteps; ++it) {
    const uint32_t cur = (uint32_t)(it & 1) * BUF_B;

    wait_async_zero();     // own DMA for buffer `cur` complete
    __syncthreads();       // everyone's DMA done AND prior reads of `nxt` done

    if (it + 1 < nsteps) { // overlap next tile DMA with this tile's WMMAs
      const uint32_t nxt = (uint32_t)((it + 1) & 1) * BUF_B;
      const int kbyte = (it + 1) * 64;          // 32 elems * 2B
      issue_tile_async<128>(Ablk, lda, kbyte, ldsb + nxt, tid);
      issue_tile_async<256>(Bblk, ldb, kbyte, ldsb + nxt + A_TILE_B, tid);
    }

    const char* smA = lds + cur;
    const char* smB = lds + cur + A_TILE_B;

    v16bf af[4], bfr[4];
    #pragma unroll
    for (int mi = 0; mi < 4; ++mi) af[mi]  = lds_a_frag(smA, mrow + mi*16, koffA);
    #pragma unroll
    for (int ni = 0; ni < 4; ++ni) bfr[ni] = lds_b_frag(smB, nrow + ni*16, koffB);

    #pragma unroll
    for (int mi = 0; mi < 4; ++mi)
      #pragma unroll
      for (int ni = 0; ni < 4; ++ni)
        acc[mi][ni] = __builtin_amdgcn_wmma_f32_16x16x32_bf16(
            false, af[mi], false, bfr[ni], (short)0, acc[mi][ni], false, false);
  }

  // Epilogue: C/D v8f layout — M = r + 8*(lane>>4), N = lane&15.
  const int lr = lane >> 4;
  const int lc = lane & 15;
  #pragma unroll
  for (int mi = 0; mi < 4; ++mi) {
    #pragma unroll
    for (int ni = 0; ni < 4; ++ni) {
      const int col = blockN + waveNl + ni*16 + lc;
      float bv = 0.f;
      if constexpr (BIAS) bv = bias[col];
      #pragma unroll
      for (int r = 0; r < 8; ++r) {
        const int row = blockM + waveMl + mi*16 + 8*lr + r;
        const float v = acc[mi][ni][r] * scale + bv;
        size_t idx;
        if constexpr (STORE_MODE == 0) {
          idx = (size_t)z * cStride + (size_t)row * ldc + col;
        } else if constexpr (STORE_MODE == 1) {
          // vhT[b][d][s]: b = row>>11, s = row&2047, d = col
          idx = ((size_t)(row >> 11) << 21) + ((size_t)col << 11) + (size_t)(row & 2047);
        } else {
          // permuted store: flat = i*1024+j <-> h*131072 + s*64 + dk
          const int h = col >> 6, dk = col & 63;
          const int prow = (h << 7) + (row >> 4);
          const int pcol = ((row & 15) << 6) + dk;
          idx = (size_t)z * cStride + (size_t)prow * ldc + pcol;
        }
        if constexpr (OUT_BF16) ((bf16_t*)Cout)[idx] = (bf16_t)v;
        else                    ((float*)Cout)[idx]  = v;
      }
    }
  }
}

// ---------------------------------------------------------------- softmax
__global__ __launch_bounds__(256) void softmax_row(const float* __restrict__ s,
                                                   bf16_t* __restrict__ d) {
  const int row = blockIdx.x;
  const float* src = s + (size_t)row * 2048;
  bf16_t*      dst = d + (size_t)row * 2048;
  const int tid  = threadIdx.x;
  const int wave = tid >> 5, lane = tid & 31;

  float v[8];
  float lmax = -3.4e38f;
  #pragma unroll
  for (int t = 0; t < 8; ++t) { v[t] = src[t*256 + tid]; lmax = fmaxf(lmax, v[t]); }
  #pragma unroll
  for (int off = 16; off; off >>= 1) lmax = fmaxf(lmax, __shfl_xor(lmax, off));
  __shared__ float sm[8];
  if (lane == 0) sm[wave] = lmax;
  __syncthreads();
  lmax = sm[0];
  #pragma unroll
  for (int w = 1; w < 8; ++w) lmax = fmaxf(lmax, sm[w]);

  float lsum = 0.f;
  #pragma unroll
  for (int t = 0; t < 8; ++t) { v[t] = __expf(v[t] - lmax); lsum += v[t]; }
  #pragma unroll
  for (int off = 16; off; off >>= 1) lsum += __shfl_xor(lsum, off);
  __shared__ float ss[8];
  if (lane == 0) ss[wave] = lsum;
  __syncthreads();
  lsum = 0.f;
  #pragma unroll
  for (int w = 0; w < 8; ++w) lsum += ss[w];

  const float rinv = 1.0f / lsum;
  #pragma unroll
  for (int t = 0; t < 8; ++t) dst[t*256 + tid] = (bf16_t)(v[t] * rinv);
}

// ---------------------------------------------------------------- host
extern "C" void kernel_launch(void* const* d_in, const int* in_sizes, int n_in,
                              void* d_out, int out_size, void* d_ws, size_t ws_size,
                              hipStream_t stream) {
  (void)in_sizes; (void)n_in; (void)out_size; (void)ws_size;
  const float* q  = (const float*)d_in[0];
  const float* k  = (const float*)d_in[1];
  const float* v  = (const float*)d_in[2];
  const float* Wq = (const float*)d_in[3];
  const float* bq = (const float*)d_in[4];
  const float* Wk = (const float*)d_in[5];
  const float* bk = (const float*)d_in[6];
  const float* Wv = (const float*)d_in[7];
  const float* bv = (const float*)d_in[8];
  const float* Wo = (const float*)d_in[9];
  const float* bo = (const float*)d_in[10];
  float* out = (float*)d_out;

  const int Bn = 4, S = 2048, D = 1024;
  const int MS = Bn * S;                 // 8192 flattened rows
  const size_t MB = 1024ull * 1024ull;
  char* ws = (char*)d_ws;

  // workspace layout (~152 MB; scores region reused for bf16 staging + outPerm)
  bf16_t* Wq_b   = (bf16_t*)(ws +   0*MB);
  bf16_t* Wk_b   = (bf16_t*)(ws +   2*MB);
  bf16_t* Wv_b   = (bf16_t*)(ws +   4*MB);
  bf16_t* Wo_b   = (bf16_t*)(ws +   6*MB);
  bf16_t* qh     = (bf16_t*)(ws +   8*MB);   // 16 MB [b][s][d]
  bf16_t* kh     = (bf16_t*)(ws +  24*MB);   // 16 MB [b][s][d]
  bf16_t* vhT    = (bf16_t*)(ws +  40*MB);   // 16 MB [b][d][s]
  float*  scores = (float*) (ws +  56*MB);   // 64 MB [b][i][j]  (K2->K3)
  bf16_t* qstg   = (bf16_t*)(ws +  56*MB);   // staging, dead before K2
  bf16_t* kstg   = (bf16_t*)(ws +  72*MB);
  bf16_t* vstg   = (bf16_t*)(ws +  88*MB);
  bf16_t* attn   = (bf16_t*)(ws + 120*MB);   // 32 MB
  bf16_t* outP   = (bf16_t*)(ws +  56*MB);   // 16 MB, reuses scores after K3

  dim3 blk(256);

  // K0: converts
  cvt_f32_to_bf16<<<2048, blk, 0, stream>>>(q, qstg, MS * D);
  cvt_f32_to_bf16<<<2048, blk, 0, stream>>>(k, kstg, MS * D);
  cvt_f32_to_bf16<<<2048, blk, 0, stream>>>(v, vstg, MS * D);
  cvt_f32_to_bf16<<< 512, blk, 0, stream>>>(Wq, Wq_b, D * D);
  cvt_f32_to_bf16<<< 512, blk, 0, stream>>>(Wk, Wk_b, D * D);
  cvt_f32_to_bf16<<< 512, blk, 0, stream>>>(Wv, Wv_b, D * D);
  cvt_f32_to_bf16<<< 512, blk, 0, stream>>>(Wo, Wo_b, D * D);

  // K1: projections (NT GEMM, 8192x1024x1024); V stored transposed
  wmma_nt_gemm<0, true, true><<<dim3(D/256, MS/128, 1), blk, 0, stream>>>(
      qstg, Wq_b, qh, bq, 1.0f, D, D, D, D, 0, 0, 0);
  wmma_nt_gemm<0, true, true><<<dim3(D/256, MS/128, 1), blk, 0, stream>>>(
      kstg, Wk_b, kh, bk, 1.0f, D, D, D, D, 0, 0, 0);
  wmma_nt_gemm<1, true, true><<<dim3(D/256, MS/128, 1), blk, 0, stream>>>(
      vstg, Wv_b, vhT, bv, 1.0f, D, D, D, 0, 0, 0, 0);

  // K2: scores = qh @ kh^T / sqrt(64), batched over B
  wmma_nt_gemm<0, false, false><<<dim3(S/256, S/128, Bn), blk, 0, stream>>>(
      qh, kh, scores, nullptr, 0.125f, D, D, D, S,
      (long)S * D, (long)S * D, (long)S * S);

  // K3: softmax
  softmax_row<<<MS, blk, 0, stream>>>(scores, attn);

  // K4: out = attn @ vh (B = vhT rows, NT), permuted bf16 store
  wmma_nt_gemm<2, true, false><<<dim3(D/256, S/128, Bn), blk, 0, stream>>>(
      attn, vhT, outP, nullptr, 1.0f, S, S, S, D,
      (long)S * S, (long)D * S, (long)S * D);

  // K5: y = outPerm @ Wo^T + bo  -> fp32
  wmma_nt_gemm<0, false, true><<<dim3(D/256, MS/128, 1), blk, 0, stream>>>(
      outP, Wo_b, out, bo, 1.0f, D, D, D, D, 0, 0, 0);
}
